// GCNLayer_90933047591262
// MI455X (gfx1250) — matrix operand order repaired and checked by reading, not compiled, not run on hardware
//
#include <hip/hip_runtime.h>
#include <hip/hip_bf16.h>
#include <stdint.h>

#define N_NODES 50000
#define N_EDGES 400000
#define IN_DIM  512
#define OUT_DIM 512

// LDS slab: 64 B-columns (W rows) x 512 K halves, padded to 520 per row
#define SLAB_ROW   520
#define SLAB_HALVES (64 * SLAB_ROW)          // one buffer, in halves

typedef _Float16 v16h __attribute__((ext_vector_type(16)));
typedef _Float16 v8h  __attribute__((ext_vector_type(8)));
typedef float    v8f  __attribute__((ext_vector_type(8)));

union Frag { v16h v; v8h h[2]; };

// ---------------- small utility kernels ----------------

__global__ void k_zero_f32(float* p, int n) {
    int i = blockIdx.x * blockDim.x + threadIdx.x;
    if (i < n) p[i] = 0.0f;
}

__global__ void k_fill_one(float* p, int n) {
    int i = blockIdx.x * blockDim.x + threadIdx.x;
    if (i < n) p[i] = 1.0f;   // self-loop contributes 1 to every node's degree
}

__global__ void k_count_deg(const int* __restrict__ src, float* __restrict__ deg) {
    int e = blockIdx.x * blockDim.x + threadIdx.x;
    if (e < N_EDGES) atomicAdd(&deg[src[e]], 1.0f);
}

__global__ void k_rsqrt(const float* __restrict__ deg, float* __restrict__ dis, int n) {
    int i = blockIdx.x * blockDim.x + threadIdx.x;
    if (i < n) dis[i] = rsqrtf(deg[i]);
}

__global__ void k_cvt_f16(const float* __restrict__ in, _Float16* __restrict__ out, int n) {
    int i = blockIdx.x * blockDim.x + threadIdx.x;
    if (i < n) out[i] = (_Float16)in[i];
}

__global__ void k_relu(float* p, int n) {
    int i = blockIdx.x * blockDim.x + threadIdx.x;
    if (i < n) p[i] = fmaxf(p[i], 0.0f);
}

// ---------------- WMMA GEMM with async-LDS double-buffered W slabs ----------------

extern __shared__ _Float16 bsl[];   // 2 x SLAB_HALVES

// Asynchronously stage W[n0..n0+63][0..511] (64 KB) into an LDS slab.
// 4096 16-byte chunks over 256 threads -> 16 async b128 loads per thread (ASYNCcnt).
__device__ __forceinline__ void stage_slab(const _Float16* __restrict__ wh, int n0,
                                           _Float16* lds_buf, int t) {
    #pragma unroll
    for (int c = 0; c < 16; ++c) {
        const int chunk = c * 256 + t;          // 0..4095
        const int row   = chunk >> 6;           // 0..63  (W row = B column)
        const int col16 = chunk & 63;           // 16B unit within the 1KB row
        const _Float16* g = wh + (size_t)(n0 + row) * IN_DIM + col16 * 8;
        const uint32_t lds = (uint32_t)(uintptr_t)(lds_buf + row * SLAB_ROW + col16 * 8);
        asm volatile("global_load_async_to_lds_b128 %0, %1, off"
                     :: "v"(lds), "v"(g)
                     : "memory");
    }
}

__global__ __launch_bounds__(256) void k_gemm_wmma(
    const _Float16* __restrict__ xh,   // [N_NODES, IN_DIM] f16
    const _Float16* __restrict__ wh,   // [OUT_DIM, IN_DIM] f16
    const float*    __restrict__ bias, // [OUT_DIM]
    _Float16*       __restrict__ h)    // [N_NODES, OUT_DIM] f16
{
    const int t    = threadIdx.x;
    const int wave = t >> 5;
    const int lane = t & 31;
    const int lo   = lane & 15;        // A: row in tile / B,C,D: column in tile
    const int hi   = lane >> 4;        // half-wave selector
    const int m0   = (blockIdx.x * 8 + wave) * 16;
    const bool active = (m0 < N_NODES);

    const _Float16* arow = xh + (size_t)(active ? (m0 + lo) : 0) * IN_DIM;

    // prologue: stage slab 0
    stage_slab(wh, 0, bsl, t);
    asm volatile("s_wait_asynccnt 0x0" ::: "memory");
    __syncthreads();

    for (int nb = 0; nb < 8; ++nb) {            // 8 blocks of 64 output columns
        const int n0 = nb * 64;
        _Float16* cur = bsl + (size_t)(nb & 1) * SLAB_HALVES;

        // async-prefetch next slab into the other buffer while we compute
        if (nb + 1 < 8)
            stage_slab(wh, (nb + 1) * 64, bsl + (size_t)((nb + 1) & 1) * SLAB_HALVES, t);

        if (active) {
            v8f acc[4];
            #pragma unroll
            for (int j = 0; j < 4; ++j) {
                const float bv = bias[n0 + j * 16 + lo];   // column bias, splat over rows
                acc[j] = (v8f){bv, bv, bv, bv, bv, bv, bv, bv};
            }

            for (int k0 = 0; k0 < IN_DIM; k0 += 32) {
                // A fragment (16x32 f16), ISA 7.12.2 layout: two contiguous 8-half runs
                Frag a;
                a.h[0] = *(const v8h*)(arow + k0 + hi * 8);
                a.h[1] = *(const v8h*)(arow + k0 + 16 + hi * 8);
                __builtin_prefetch(arow + k0 + 32, 0, 3);

                #pragma unroll
                for (int j = 0; j < 4; ++j) {
                    // B fragment: two 8-K runs (k0+hi*8, k0+16+hi*8) of column n, from LDS (bank-padded)
                    const _Float16* bp = cur + (j * 16 + lo) * SLAB_ROW + k0 + hi * 8;
                    Frag b;
                    b.h[0] = *(const v8h*)(bp);
                    b.h[1] = *(const v8h*)(bp + 16);
                    acc[j] = __builtin_amdgcn_wmma_f32_16x16x32_f16(
                        false, a.v, false, b.v, (short)0, acc[j], false, false);
                }
            }

            // D layout: component r -> row m0 + r + hi*8, col n0 + j*16 + lo
            #pragma unroll
            for (int j = 0; j < 4; ++j) {
                const int n = n0 + j * 16 + lo;
                _Float16* hp = h + (size_t)(m0 + hi * 8) * OUT_DIM + n;
                #pragma unroll
                for (int r = 0; r < 8; ++r)
                    hp[(size_t)r * OUT_DIM] = (_Float16)acc[j][r];
            }
        }

        // my staging of the next buffer is done; barrier publishes it block-wide
        asm volatile("s_wait_asynccnt 0x0" ::: "memory");
        __syncthreads();
    }
}

// ---------------- scatter: out[dst] += dis[src]*dis[dst]*h[src] ----------------
// Wave per (virtual) edge; edges [N_EDGES, N_EDGES+N_NODES) are self loops.

__global__ __launch_bounds__(256) void k_scatter(
    const int* __restrict__ ei,
    const float* __restrict__ dis,
    const _Float16* __restrict__ h,
    float* __restrict__ out)
{
    const int wave = threadIdx.x >> 5;
    const int lane = threadIdx.x & 31;
    const long long e = (long long)blockIdx.x * 8 + wave;
    if (e >= (long long)N_EDGES + N_NODES) return;

    int src, dst;
    if (e < N_EDGES) { src = ei[e]; dst = ei[N_EDGES + e]; }
    else             { src = dst = (int)(e - N_EDGES); }

    const float norm = dis[src] * dis[dst];

    const _Float16* hs = h   + (size_t)src * OUT_DIM + lane * 16;
    float*          op = out + (size_t)dst * OUT_DIM + lane * 16;

    const v8h h0 = *(const v8h*)(hs);
    const v8h h1 = *(const v8h*)(hs + 8);

    #pragma unroll
    for (int i = 0; i < 8; ++i) atomicAdd(op + i,     norm * (float)h0[i]);
    #pragma unroll
    for (int i = 0; i < 8; ++i) atomicAdd(op + 8 + i, norm * (float)h1[i]);
}

// ---------------- launcher ----------------

extern "C" void kernel_launch(void* const* d_in, const int* in_sizes, int n_in,
                              void* d_out, int out_size, void* d_ws, size_t ws_size,
                              hipStream_t stream) {
    const float* x  = (const float*)d_in[0];   // [N_NODES, IN_DIM]
    const int*   ei = (const int*)d_in[1];     // [2, N_EDGES]
    const float* W  = (const float*)d_in[2];   // [OUT_DIM, IN_DIM]
    const float* b  = (const float*)d_in[3];   // [OUT_DIM]
    float* out = (float*)d_out;                // [N_NODES, OUT_DIM]

    char* ws = (char*)d_ws;
    _Float16* xh  = (_Float16*)(ws);                        // 51,200,000 B
    _Float16* wh  = (_Float16*)(ws + 51200000);             //    524,288 B
    _Float16* h   = (_Float16*)(ws + 51724288);             // 51,200,000 B
    float*    deg = (float*)   (ws + 102924288);            //    200,000 B
    float*    dis = (float*)   (ws + 103124288);            //    200,000 B

    const int T = 256;
    const int n_out = N_NODES * OUT_DIM;   // 25,600,000
    const int n_x   = N_NODES * IN_DIM;    // 25,600,000
    const int n_w   = OUT_DIM * IN_DIM;    //    262,144

    k_zero_f32<<<(n_out + T - 1) / T, T, 0, stream>>>(out, n_out);
    k_fill_one<<<(N_NODES + T - 1) / T, T, 0, stream>>>(deg, N_NODES);
    k_count_deg<<<(N_EDGES + T - 1) / T, T, 0, stream>>>(ei, deg);
    k_rsqrt<<<(N_NODES + T - 1) / T, T, 0, stream>>>(deg, dis, N_NODES);
    k_cvt_f16<<<(n_x + T - 1) / T, T, 0, stream>>>(x, xh, n_x);
    k_cvt_f16<<<(n_w + T - 1) / T, T, 0, stream>>>(W, wh, n_w);

    const int m_tiles = N_NODES / 16;                 // 3125
    const size_t lds_bytes = 2u * SLAB_HALVES * sizeof(_Float16);   // 133,120 B
    k_gemm_wmma<<<(m_tiles + 7) / 8, T, lds_bytes, stream>>>(xh, wh, b, h);

    const long long n_virt_edges = (long long)N_EDGES + N_NODES;    // 450,000
    k_scatter<<<(unsigned)((n_virt_edges + 7) / 8), T, 0, stream>>>(ei, dis, h, out);

    k_relu<<<(n_out + T - 1) / T, T, 0, stream>>>(out, n_out);
}
